// ClimateGNN_3667902070929
// MI455X (gfx1250) — compile-verified, matrix operand order checked by reading, not run on hardware
//
#include <hip/hip_runtime.h>
#include <hip/hip_bf16.h>

// ---------------------------------------------------------------------------
// ClimateGNN (3-layer GAT + BN + pool + MLP) for MI455X (gfx1250, wave32).
//
// GEMMs run on v_wmma_f32_16x16x32_bf16 (bf16 in, f32 accumulate; every layer
// is batch-normed immediately after, so bf16 inputs are adequate). Each wave
// computes a 16x64 macro-tile (4 accumulators) so the A fragment is reused
// 4x -> 2.5 b128 loads per WMMA instead of 4. Edge phase (the real bottleneck)
// keeps xW (102MB) resident in the 192MB L2 and uses f32 atomics through L2.
// Biases before BN cancel exactly and are dropped.
// ---------------------------------------------------------------------------

#define N_NODES 50000
#define E_EDGES 400000
#define ETOT    (E_EDGES + N_NODES)
#define HIDDEN  128

typedef __bf16 bf16_t;
typedef __attribute__((ext_vector_type(8)))  __bf16 v8bf;
typedef __attribute__((ext_vector_type(16))) __bf16 v16bf;
typedef __attribute__((ext_vector_type(8)))  float  v8f;

union V16 { v16bf v; v8bf h[2]; };

// ---------------- small utility kernels ----------------

__global__ void fill_f32(float* p, float v, int n) {
    int i = blockIdx.x * blockDim.x + threadIdx.x;
    if (i < n) p[i] = v;
}

__global__ void fill_u32(unsigned* p, unsigned v, int n) {
    int i = blockIdx.x * blockDim.x + threadIdx.x;
    if (i < n) p[i] = v;
}

__global__ void cast_f32_bf16(const float* __restrict__ s, bf16_t* __restrict__ d, int n) {
    int i = blockIdx.x * blockDim.x + threadIdx.x;
    if (i < n) d[i] = (bf16_t)s[i];
}

// W [Fin, H, C] (== [K, Cout] flat) -> Wt [Cout, K] bf16 (so B-fragment loads
// are contiguous, mirroring the A-fragment layout).
__global__ void transpose_cast_W(const float* __restrict__ W, bf16_t* __restrict__ Wt,
                                 int K, int Cout) {
    int i = blockIdx.x * blockDim.x + threadIdx.x;
    if (i >= K * Cout) return;
    int f  = i / Cout;
    int co = i % Cout;
    Wt[co * K + f] = (bf16_t)W[i];
}

// ---------------- WMMA GEMM: C[N,Cout] = A[N,K](bf16) x Bt[Cout,K](bf16) ----

// One wave computes a 16x64 macro-tile: 4 WMMA accumulators sharing one A
// fragment per K-step (A loaded once, used 4x).
// 16-bit A/B fragment layout (ISA 7.12.2): lane L holds row/col (L&15);
// K-halves [0..7]|[16..23] for lanes 0-15, [8..15]|[24..31] for lanes 16-31.
__global__ __launch_bounds__(256) void gemm_wmma_bf16(
    const bf16_t* __restrict__ A, const bf16_t* __restrict__ Bt,
    float* __restrict__ C, int K, int Cout, int nTiles) {
    int wid  = (blockIdx.x * blockDim.x + threadIdx.x) >> 5;
    int lane = threadIdx.x & 31;
    if (wid >= nTiles) return;

    int nCG     = Cout >> 6;          // 64-wide column groups (8 or 2)
    int rowTile = wid / nCG;          // 0..3124  (N/16 = 3125 exactly)
    int cg      = wid % nCG;          // covers cols [cg*64, cg*64+63]

    int row   = rowTile * 16 + (lane & 15);
    int khalf = (lane >> 4) * 8;

    const bf16_t* ap = A  + (size_t)row * K + khalf;
    const bf16_t* bp = Bt + (size_t)(cg * 64 + (lane & 15)) * K + khalf;

    v8f acc0 = {}, acc1 = {}, acc2 = {}, acc3 = {};
    for (int k0 = 0; k0 < K; k0 += 32) {
        V16 a, b0, b1, b2, b3;
        a.h[0]  = *(const v8bf*)(ap + k0);
        a.h[1]  = *(const v8bf*)(ap + k0 + 16);
        b0.h[0] = *(const v8bf*)(bp + k0);
        b0.h[1] = *(const v8bf*)(bp + k0 + 16);
        b1.h[0] = *(const v8bf*)(bp + 16 * K + k0);
        b1.h[1] = *(const v8bf*)(bp + 16 * K + k0 + 16);
        b2.h[0] = *(const v8bf*)(bp + 32 * K + k0);
        b2.h[1] = *(const v8bf*)(bp + 32 * K + k0 + 16);
        b3.h[0] = *(const v8bf*)(bp + 48 * K + k0);
        b3.h[1] = *(const v8bf*)(bp + 48 * K + k0 + 16);
        acc0 = __builtin_amdgcn_wmma_f32_16x16x32_bf16(false, a.v, false, b0.v, (short)0, acc0, false, false);
        acc1 = __builtin_amdgcn_wmma_f32_16x16x32_bf16(false, a.v, false, b1.v, (short)0, acc1, false, false);
        acc2 = __builtin_amdgcn_wmma_f32_16x16x32_bf16(false, a.v, false, b2.v, (short)0, acc2, false, false);
        acc3 = __builtin_amdgcn_wmma_f32_16x16x32_bf16(false, a.v, false, b3.v, (short)0, acc3, false, false);
    }

    // C/D layout: VGPR r -> M = r (+8 for lanes 16-31); N = lane&15.
    int    m    = rowTile * 16 + ((lane >> 4) ? 8 : 0);
    float* base = C + (size_t)m * Cout + cg * 64 + (lane & 15);
#pragma unroll
    for (int r = 0; r < 8; ++r) {
        float* cp = base + (size_t)r * Cout;
        cp[0]  = acc0[r];
        cp[16] = acc1[r];
        cp[32] = acc2[r];
        cp[48] = acc3[r];
    }
}

// ---------------- attention logits ----------------

// i = n*H + h indexes directly into xW rows [N*H, 128].
__global__ void attn_logits(const float* __restrict__ xW,
                            const float* __restrict__ asrc,
                            const float* __restrict__ adst,
                            float* __restrict__ als, float* __restrict__ ald, int H) {
    int i = blockIdx.x * blockDim.x + threadIdx.x;
    if (i >= N_NODES * H) return;
    int h = i % H;
    const float* v  = xW   + (size_t)i * HIDDEN;
    const float* as = asrc + h * HIDDEN;
    const float* ad = adst + h * HIDDEN;
    float s = 0.f, d = 0.f;
#pragma unroll 4
    for (int c = 0; c < HIDDEN; ++c) { float x = v[c]; s += x * as[c]; d += x * ad[c]; }
    als[i] = s;
    ald[i] = d;
}

// ---------------- segment softmax over edges ----------------

__device__ __forceinline__ unsigned f2ord(float f) {
    unsigned u = __float_as_uint(f);
    return (u >> 31) ? ~u : (u | 0x80000000u);
}
__device__ __forceinline__ float ord2f(unsigned o) {
    return __uint_as_float((o >> 31) ? (o & 0x7fffffffu) : ~o);
}

__device__ __forceinline__ void edge_sd(const int* ei, int e, int& s, int& d) {
    if (e < E_EDGES) { s = ei[e]; d = ei[E_EDGES + e]; }
    else             { s = d = e - E_EDGES; }   // self loops
}

// pass 1: e = leaky_relu(als[src]+ald[dst]); per-dst max via ordered-uint atomicMax
__global__ void edge_pass1(const int* __restrict__ ei,
                           const float* __restrict__ als, const float* __restrict__ ald,
                           float* __restrict__ ebuf, unsigned* __restrict__ mord, int H) {
    int i = blockIdx.x * blockDim.x + threadIdx.x;
    if (i >= ETOT * H) return;
    int e = i / H, h = i % H;
    int s, d; edge_sd(ei, e, s, d);
    float t = als[s * H + h] + ald[d * H + h];
    t = (t > 0.f) ? t : 0.2f * t;              // leaky_relu, NEG_SLOPE=0.2
    ebuf[i] = t;
    atomicMax(mord + d * H + h, f2ord(t));
}

// pass 2: ex = exp(e - max); per-dst sum
__global__ void edge_pass2(const int* __restrict__ ei,
                           float* __restrict__ ebuf, const unsigned* __restrict__ mord,
                           float* __restrict__ ssum, int H) {
    int i = blockIdx.x * blockDim.x + threadIdx.x;
    if (i >= ETOT * H) return;
    int e = i / H, h = i % H;
    int s, d; edge_sd(ei, e, s, d);
    float m  = ord2f(mord[d * H + h]);
    float ex = __expf(ebuf[i] - m);
    ebuf[i]  = ex;
    atomicAdd(ssum + d * H + h, ex);
}

// pass 3: agg[dst,c] += (1/H) * sum_h alpha_h * xW[src,h,c]  (wave per edge)
__global__ __launch_bounds__(256) void edge_aggregate(
    const int* __restrict__ ei, const float* __restrict__ xW,
    const float* __restrict__ ebuf, const float* __restrict__ ssum,
    float* __restrict__ agg, int H) {
    int e    = (blockIdx.x * blockDim.x + threadIdx.x) >> 5;
    int lane = threadIdx.x & 31;
    if (e >= ETOT) return;
    int s, d; edge_sd(ei, e, s, d);

    float invH = 1.0f / (float)H;
    float w[4];
    for (int h = 0; h < H; ++h)
        w[h] = ebuf[e * H + h] / (ssum[d * H + h] + 1e-16f) * invH;

    const float* xr = xW + (size_t)s * H * HIDDEN;
    float*       ar = agg + (size_t)d * HIDDEN;
    __builtin_prefetch(xr, 0, 1);              // global_prefetch_b8 into L2
#pragma unroll
    for (int cc = 0; cc < HIDDEN; cc += 32) {
        int   c   = cc + lane;
        float acc = 0.f;
        for (int h = 0; h < H; ++h) acc += w[h] * xr[h * HIDDEN + c];
        atomicAdd(ar + c, acc);
    }
}

// ---------------- BN stats / pooling / apply / MLP ----------------

// block per feature column; computes column mean (and rsqrt(var+eps) if wantVar)
__global__ __launch_bounds__(256) void column_stats(const float* __restrict__ x,
                                                    float* __restrict__ mu,
                                                    float* __restrict__ rv,
                                                    int n, int wantVar) {
    int c = blockIdx.x;                        // 0..127
    float s = 0.f, q = 0.f;
    for (int r = threadIdx.x; r < n; r += blockDim.x) {
        float v = x[(size_t)r * HIDDEN + c];
        s += v; q += v * v;
    }
    __shared__ float sh[256], qh[256];
    sh[threadIdx.x] = s; qh[threadIdx.x] = q;
    __syncthreads();
    for (int off = 128; off > 0; off >>= 1) {
        if (threadIdx.x < off) {
            sh[threadIdx.x] += sh[threadIdx.x + off];
            qh[threadIdx.x] += qh[threadIdx.x + off];
        }
        __syncthreads();
    }
    if (threadIdx.x == 0) {
        float m = sh[0] / (float)n;
        mu[c] = m;
        if (wantVar) {
            float var = qh[0] / (float)n - m * m;
            rv[c] = rsqrtf(var + 1e-5f);
        }
    }
}

__global__ void bn_apply(const float* __restrict__ in, float* __restrict__ out,
                         const float* __restrict__ mu, const float* __restrict__ rv,
                         const float* __restrict__ gamma, const float* __restrict__ beta,
                         int relu) {
    int i = blockIdx.x * blockDim.x + threadIdx.x;
    if (i >= N_NODES * HIDDEN) return;
    int c = i & (HIDDEN - 1);
    float v = (in[i] - mu[c]) * rv[c] * gamma[c] + beta[c];
    if (relu) v = fmaxf(v, 0.f);
    out[i] = v;
}

__global__ __launch_bounds__(64) void mlp_head(const float* __restrict__ pooled,
                                               const float* __restrict__ pW1,
                                               const float* __restrict__ pb1,
                                               const float* __restrict__ pW2,
                                               const float* __restrict__ pb2,
                                               float* __restrict__ out) {
    __shared__ float hid[64];
    int j = threadIdx.x;
    float a = pb1[j];
#pragma unroll 4
    for (int c = 0; c < HIDDEN; ++c) a += pooled[c] * pW1[c * 64 + j];
    hid[j] = fmaxf(a, 0.f);
    __syncthreads();
    if (j == 0) {
        float o = pb2[0];
        for (int jj = 0; jj < 64; ++jj) o += hid[jj] * pW2[jj];
        out[0] = o;
    }
}

// ---------------- host orchestration ----------------

static inline int ceil_div(int a, int b) { return (a + b - 1) / b; }

// f2ord(-inf) = ~0xFF800000 = 0x007FFFFF  (ordered-uint encoding of -infinity)
#define ORD_NEG_INF 0x007FFFFFu

extern "C" void kernel_launch(void* const* d_in, const int* in_sizes, int n_in,
                              void* d_out, int out_size, void* d_ws, size_t ws_size,
                              hipStream_t stream) {
    (void)in_sizes; (void)n_in; (void)out_size; (void)ws_size;

    const float* x  = (const float*)d_in[0];
    const int*   ei = (const int*)d_in[1];
    const float* W[3]     = { (const float*)d_in[2],  (const float*)d_in[8],  (const float*)d_in[14] };
    const float* asrc[3]  = { (const float*)d_in[3],  (const float*)d_in[9],  (const float*)d_in[15] };
    const float* adst[3]  = { (const float*)d_in[4],  (const float*)d_in[10], (const float*)d_in[16] };
    // bias{0,1,2} (d_in[5,11,17]) cancel exactly under the following BN -> dropped.
    const float* gamma[3] = { (const float*)d_in[6],  (const float*)d_in[12], (const float*)d_in[18] };
    const float* beta[3]  = { (const float*)d_in[7],  (const float*)d_in[13], (const float*)d_in[19] };
    const float* pW1 = (const float*)d_in[20];
    const float* pb1 = (const float*)d_in[21];
    const float* pW2 = (const float*)d_in[22];
    const float* pb2 = (const float*)d_in[23];
    float* out = (float*)d_out;

    // workspace bump allocator (256B aligned); total ~177 MB
    char* wp = (char*)d_ws;
    auto alloc = [&](size_t bytes) -> char* {
        char* p = wp;
        wp += (bytes + 255) & ~(size_t)255;
        return p;
    };
    bf16_t*   xb   = (bf16_t*)alloc((size_t)N_NODES * 128 * 2);   // bf16 activations
    bf16_t*   Wt   = (bf16_t*)alloc((size_t)512 * 128 * 2);       // bf16 W^T [Cout,K]
    float*    xW   = (float*) alloc((size_t)N_NODES * 512 * 4);   // GEMM out [N,H,128]
    float*    agg  = (float*) alloc((size_t)N_NODES * 128 * 4);   // aggregation
    float*    hbuf = (float*) alloc((size_t)N_NODES * 128 * 4);   // layer output
    float*    als  = (float*) alloc((size_t)N_NODES * 4 * 4);
    float*    ald  = (float*) alloc((size_t)N_NODES * 4 * 4);
    unsigned* mord = (unsigned*)alloc((size_t)N_NODES * 4 * 4);
    float*    ssum = (float*) alloc((size_t)N_NODES * 4 * 4);
    float*    ebuf = (float*) alloc((size_t)ETOT * 4 * 4);
    float*    mu   = (float*) alloc(128 * 4);
    float*    rv   = (float*) alloc(128 * 4);
    float*    pooled = (float*)alloc(128 * 4);

    const int Hs[3]  = { 4, 4, 1 };
    const int Kin[3] = { 64, 128, 128 };
    const float* cur = x;

    for (int L = 0; L < 3; ++L) {
        int K = Kin[L], H = Hs[L], Cout = H * HIDDEN;

        // cast + pack operands for WMMA
        cast_f32_bf16<<<ceil_div(N_NODES * K, 256), 256, 0, stream>>>(cur, xb, N_NODES * K);
        transpose_cast_W<<<ceil_div(K * Cout, 256), 256, 0, stream>>>(W[L], Wt, K, Cout);

        // xW = X x W  (WMMA bf16 -> f32); one wave per 16x64 macro-tile
        int nTiles = (N_NODES / 16) * (Cout / 64);
        gemm_wmma_bf16<<<ceil_div(nTiles, 8), 256, 0, stream>>>(xb, Wt, xW, K, Cout, nTiles);

        // per-node attention logits
        attn_logits<<<ceil_div(N_NODES * H, 256), 256, 0, stream>>>(xW, asrc[L], adst[L], als, ald, H);

        // init segment-softmax state (ws is poisoned by harness -> must reset every call)
        fill_u32<<<ceil_div(N_NODES * H, 256), 256, 0, stream>>>(mord, ORD_NEG_INF, N_NODES * H);
        fill_f32<<<ceil_div(N_NODES * H, 256), 256, 0, stream>>>(ssum, 0.f, N_NODES * H);
        fill_f32<<<ceil_div(N_NODES * HIDDEN, 256), 256, 0, stream>>>(agg, 0.f, N_NODES * HIDDEN);

        edge_pass1<<<ceil_div(ETOT * H, 256), 256, 0, stream>>>(ei, als, ald, ebuf, mord, H);
        edge_pass2<<<ceil_div(ETOT * H, 256), 256, 0, stream>>>(ei, ebuf, mord, ssum, H);
        edge_aggregate<<<ceil_div(ETOT * 32, 256), 256, 0, stream>>>(ei, xW, ebuf, ssum, agg, H);

        // BN (+ReLU for layers 0,1)
        column_stats<<<128, 256, 0, stream>>>(agg, mu, rv, N_NODES, 1);
        bn_apply<<<ceil_div(N_NODES * HIDDEN, 256), 256, 0, stream>>>(
            agg, hbuf, mu, rv, gamma[L], beta[L], (L < 2) ? 1 : 0);
        cur = hbuf;
    }

    // mean pool over nodes, then MLP head -> scalar
    column_stats<<<128, 256, 0, stream>>>(hbuf, pooled, rv, N_NODES, 0);
    mlp_head<<<1, 64, 0, stream>>>(pooled, pW1, pb1, pW2, pb2, out);
}
